// BitchyNetwork_23940147708524
// MI455X (gfx1250) — matrix operand order repaired and checked by reading, not compiled, not run on hardware
//
#include <hip/hip_runtime.h>
#include <hip/hip_bf16.h>
#include <cstdint>
#include <cstddef>

// Problem constants (match reference).
#define N_ 4096
#define D_ 512
#define F_ 16384

typedef __attribute__((ext_vector_type(16))) __bf16 v16bf;
typedef __attribute__((ext_vector_type(8)))  __bf16 v8bf;
typedef __attribute__((ext_vector_type(8)))  float  v8f;

union ABfrag { v16bf v; v8bf h[2]; };

// ---------------------------------------------------------------------------
// Kernel A: raw[n][f] = (x[n]-b_dec) . W_enc[f] + b_enc[f]
// bf16x3 split-precision WMMA GEMM (hi*hi + hi*lo + lo*hi), fp32 accumulate.
// Tile: 64 rows (M) x 64 features (F) per 128-thread (4-wave) workgroup.
// Each wave owns a 16-row strip and 4 accumulator tiles (16x16 each).
// ---------------------------------------------------------------------------
#define TM 64
#define TF 64
#define TK 32

__global__ __launch_bounds__(128) void enc_gemm_wmma(
    const float* __restrict__ x, const float* __restrict__ Wenc,
    const float* __restrict__ b_enc, const float* __restrict__ b_dec,
    float* __restrict__ raw)
{
  __shared__ __bf16 sxh[TM][TK];   // x tile, bf16 high part
  __shared__ __bf16 sxl[TM][TK];   // x tile, bf16 residual
  __shared__ __bf16 swh[TF][TK];   // W tile, bf16 high part
  __shared__ __bf16 swl[TF][TK];   // W tile, bf16 residual

  const int t    = threadIdx.x;
  const int lane = t & 31;
  const int wave = t >> 5;
  const int m0   = blockIdx.y * TM;
  const int f0   = blockIdx.x * TF;

  // Staging assignment: 2 threads per 32-wide row, 16 contiguous K floats each.
  const int trow = t >> 1;
  const int tkh  = (t & 1) * 16;

  v8f acc[4];
#pragma unroll
  for (int i = 0; i < 4; ++i)
#pragma unroll
    for (int j = 0; j < 8; ++j) acc[i][j] = 0.0f;

  for (int kk = 0; kk < D_; kk += TK) {
    __syncthreads();   // protect previous iteration's LDS reads

    const float* gx = x    + (size_t)(m0 + trow) * D_ + kk + tkh;
    const float* gw = Wenc + (size_t)(f0 + trow) * D_ + kk + tkh;
    if (kk + TK < D_) {                       // gfx1250 global_prefetch_b8
      __builtin_prefetch(gx + TK, 0, 1);
      __builtin_prefetch(gw + TK, 0, 1);
    }
#pragma unroll
    for (int i = 0; i < 16; i += 4) {
      float4 qx = *(const float4*)(gx + i);
      float4 qb = *(const float4*)(b_dec + kk + tkh + i);
      float vv[4] = {qx.x - qb.x, qx.y - qb.y, qx.z - qb.z, qx.w - qb.w};
#pragma unroll
      for (int c = 0; c < 4; ++c) {
        __bf16 hi = (__bf16)vv[c];
        __bf16 lo = (__bf16)(vv[c] - (float)hi);
        sxh[trow][tkh + i + c] = hi;
        sxl[trow][tkh + i + c] = lo;
      }
      float4 qw = *(const float4*)(gw + i);
      float ww[4] = {qw.x, qw.y, qw.z, qw.w};
#pragma unroll
      for (int c = 0; c < 4; ++c) {
        __bf16 hi = (__bf16)ww[c];
        __bf16 lo = (__bf16)(ww[c] - (float)hi);
        swh[trow][tkh + i + c] = hi;
        swl[trow][tkh + i + c] = lo;
      }
    }
    __syncthreads();

    // A fragment (16x32 bf16): lane l holds row m = l&15;
    // halves [0..7] = K {hl*16..+7}, halves [8..15] = K {hl*16+8..+7}.
    {
      const int r  = lane & 15;
      const int hl = lane >> 4;
      ABfrag ah, al;
      const __bf16* pax = &sxh[wave * 16 + r][hl * 16];
      ah.h[0] = *(const v8bf*)pax;
      ah.h[1] = *(const v8bf*)(pax + 8);
      const __bf16* pal = &sxl[wave * 16 + r][hl * 16];
      al.h[0] = *(const v8bf*)pal;
      al.h[1] = *(const v8bf*)(pal + 8);

      // B fragment (32x16 bf16): lane l holds column n = l&15, K = 16*hl..+15
      // contiguous -> a contiguous 32B row slice of the W tile.
#pragma unroll
      for (int fs = 0; fs < 4; ++fs) {
        ABfrag bh, bl;
        bh.v = *(const v16bf*)&swh[fs * 16 + r][hl * 16];
        bl.v = *(const v16bf*)&swl[fs * 16 + r][hl * 16];
        acc[fs] = __builtin_amdgcn_wmma_f32_16x16x32_bf16(
            false, ah.v, false, bh.v, (short)0, acc[fs], false, false);
        acc[fs] = __builtin_amdgcn_wmma_f32_16x16x32_bf16(
            false, ah.v, false, bl.v, (short)0, acc[fs], false, false);
        acc[fs] = __builtin_amdgcn_wmma_f32_16x16x32_bf16(
            false, al.v, false, bh.v, (short)0, acc[fs], false, false);
      }
    }
  }

  // Epilogue. C/D layout: VGPR j -> M = j + 8*(lane>=16), N = lane&15.
  {
    const int r  = lane & 15;
    const int hl = lane >> 4;
#pragma unroll
    for (int fs = 0; fs < 4; ++fs) {
      const int   fcol = f0 + fs * 16 + r;
      const float be   = b_enc[fcol];
#pragma unroll
      for (int j = 0; j < 8; ++j) {
        const int m = m0 + wave * 16 + j + 8 * hl;
        raw[(size_t)m * F_ + fcol] = acc[fs][j] + be;
      }
    }
  }
}

// ---------------------------------------------------------------------------
// Kernel B: exact K-th largest per row via 4-pass radix select on the
// sign-flipped orderable uint representation. One 256-thread block per row.
// ---------------------------------------------------------------------------
__global__ __launch_bounds__(256) void topk_threshold(
    const float* __restrict__ raw, const int* __restrict__ numw,
    float* __restrict__ thresh)
{
  __shared__ unsigned hist[256];
  __shared__ unsigned s_prefix;
  __shared__ int      s_k;
  const int tid = threadIdx.x;
  const int row = blockIdx.x;
  const float* r = raw + (size_t)row * F_;

  if (tid == 0) { s_prefix = 0u; s_k = numw[0]; }

  for (int pass = 0; pass < 4; ++pass) {
    hist[tid] = 0u;
    __syncthreads();
    const int      sh    = 24 - 8 * pass;
    const unsigned pmask = pass ? (0xFFFFFFFFu << (sh + 8)) : 0u;
    const unsigned pref  = s_prefix;
    for (int i = tid; i < F_; i += 256) {
      unsigned u = __float_as_uint(r[i]);
      u = (u & 0x80000000u) ? ~u : (u | 0x80000000u);
      if ((u & pmask) == pref) atomicAdd(&hist[(u >> sh) & 255u], 1u);
    }
    __syncthreads();
    if (tid == 0) {
      int kk = s_k;
      int d  = 255;
      while (d > 0 && (int)hist[d] < kk) { kk -= (int)hist[d]; --d; }
      s_prefix = pref | ((unsigned)d << sh);
      s_k      = kk;
    }
    __syncthreads();
  }
  if (tid == 0) {
    const unsigned u    = s_prefix;
    const unsigned bits = (u & 0x80000000u) ? (u & 0x7FFFFFFFu) : ~u;
    thresh[row] = __uint_as_float(bits);
  }
}

// ---------------------------------------------------------------------------
// Kernel C: mask acts in place, sparse decode (<=64 winners), per-row sq err.
// Winners are sorted by feature index so FP accumulation order is fixed
// (bit-deterministic across replays).
// ---------------------------------------------------------------------------
__global__ __launch_bounds__(256) void mask_recon(
    float* __restrict__ acts, const float* __restrict__ thresh,
    const float* __restrict__ Wdec, const float* __restrict__ b_dec,
    const float* __restrict__ x, float* __restrict__ recon,
    float* __restrict__ rowerr)
{
  __shared__ float s_val[64];
  __shared__ int   s_f[64];
  __shared__ int   s_cnt;
  __shared__ float red[256];
  const int tid = threadIdx.x;
  const int row = blockIdx.x;
  float* arow = acts + (size_t)row * F_;
  const float th = thresh[row];

  if (tid == 0) s_cnt = 0;
  __syncthreads();

  for (int i = tid; i < F_; i += 256) {
    const float v  = arow[i];
    const bool win = (v >= th);
    if (win) {
      const int slot = atomicAdd(&s_cnt, 1);
      if (slot < 64) { s_val[slot] = v; s_f[slot] = i; }
    }
    arow[i] = win ? v : 0.0f;
  }
  __syncthreads();

  const int nw = (s_cnt < 64) ? s_cnt : 64;
  if (tid == 0) {   // insertion sort by feature index (nw <= 64)
    for (int a = 1; a < nw; ++a) {
      const int fa = s_f[a];
      const float va = s_val[a];
      int b = a - 1;
      while (b >= 0 && s_f[b] > fa) {
        s_f[b + 1] = s_f[b]; s_val[b + 1] = s_val[b]; --b;
      }
      s_f[b + 1] = fa; s_val[b + 1] = va;
    }
  }
  __syncthreads();

  float err = 0.0f;
  for (int d = tid; d < D_; d += 256) {
    float rec = b_dec[d];
    for (int j = 0; j < nw; ++j)
      rec = fmaf(s_val[j], Wdec[(size_t)s_f[j] * D_ + d], rec);
    recon[(size_t)row * D_ + d] = rec;
    const float df = rec - x[(size_t)row * D_ + d];
    err = fmaf(df, df, err);
  }
  red[tid] = err;
  __syncthreads();
  for (int s = 128; s > 0; s >>= 1) {
    if (tid < s) red[tid] += red[tid + s];
    __syncthreads();
  }
  if (tid == 0) rowerr[row] = red[0];
}

// ---------------------------------------------------------------------------
// Kernel D: deterministic final loss reduction.
// ---------------------------------------------------------------------------
__global__ __launch_bounds__(256) void reduce_loss(
    const float* __restrict__ rowerr, float* __restrict__ loss)
{
  __shared__ float red[256];
  const int tid = threadIdx.x;
  float s = 0.0f;
  for (int i = tid; i < N_; i += 256) s += rowerr[i];
  red[tid] = s;
  __syncthreads();
  for (int st = 128; st > 0; st >>= 1) {
    if (tid < st) red[tid] += red[tid + st];
    __syncthreads();
  }
  if (tid == 0) *loss = red[0] * (1.0f / (float)N_);
}

// ---------------------------------------------------------------------------
extern "C" void kernel_launch(void* const* d_in, const int* in_sizes, int n_in,
                              void* d_out, int out_size, void* d_ws, size_t ws_size,
                              hipStream_t stream) {
  (void)in_sizes; (void)n_in; (void)out_size; (void)ws_size;

  const float* x     = (const float*)d_in[0];
  const float* W_enc = (const float*)d_in[1];
  const float* W_dec = (const float*)d_in[2];
  const float* b_enc = (const float*)d_in[3];
  const float* b_dec = (const float*)d_in[4];
  const int*   numw  = (const int*)d_in[5];

  float* out   = (float*)d_out;
  float* loss  = out;                          // [1]
  float* recon = out + 1;                      // [N*D]
  float* acts  = out + 1 + (size_t)N_ * D_;    // [N*F]; raw logits live here
                                               // between kernels A and C.
  float* thresh = (float*)d_ws;                // [N]
  float* rowerr = thresh + N_;                 // [N]

  dim3 gA(F_ / TF, N_ / TM);
  enc_gemm_wmma<<<gA, 128, 0, stream>>>(x, W_enc, b_enc, b_dec, acts);
  topk_threshold<<<N_, 256, 0, stream>>>(acts, numw, thresh);
  mask_recon<<<N_, 256, 0, stream>>>(acts, thresh, W_dec, b_dec, x, recon, rowerr);
  reduce_loss<<<1, 256, 0, stream>>>(rowerr, loss);
}
